// YemongFull_807453851685
// MI455X (gfx1250) — matrix operand order repaired and checked by reading, not compiled, not run on hardware
//
#include <hip/hip_runtime.h>
#include <math.h>

// ---------------- model constants ----------------
#define BB 8
#define SEQ 256
#define NS 8
#define DM 512
#define DI 1024
#define DSTATE 16
#define DTRANK 32
#define NH 8
#define DH 64
#define BT_TOK (BB*SEQ*NS)       /* 16384 tokens, layout [B,S,N,*] */
#define BSNN  (BB*SEQ*NS*NS)     /* 131072 pair rows */
#define CDIV(a,b) (((a)+(b)-1)/(b))

typedef __attribute__((ext_vector_type(16))) __bf16        v16bf;
typedef __attribute__((ext_vector_type(8)))  float         v8f;
typedef __attribute__((ext_vector_type(4)))  float         f32x4;
typedef __attribute__((ext_vector_type(4)))  unsigned int  u32x4;

union Frag { v16bf v; u32x4 q[2]; };

__device__ __forceinline__ unsigned int f2bf(float f) {
  union { float f; unsigned int u; } c; c.f = f;
  unsigned int u = c.u;
  unsigned int r = u + 0x7FFFu + ((u >> 16) & 1u);      // round-to-nearest-even
  if ((u & 0x7F800000u) == 0x7F800000u) r = u;           // inf/nan passthrough
  return r >> 16;
}
__device__ __forceinline__ unsigned int pack2(float lo, float hi) {
  return f2bf(lo) | (f2bf(hi) << 16);
}
__device__ __forceinline__ float siluf(float v){ return v / (1.f + __expf(-v)); }

enum { GF_BIAS=1, GF_ACCUM=2, GF_SILU=4, GF_SOFTPLUS=8 };

// ---------------- TDM (Tensor Data Mover) support ----------------
#if __has_builtin(__builtin_amdgcn_tensor_load_to_lds) && __has_builtin(__builtin_amdgcn_s_wait_tensorcnt)
#define USE_TDM 1
typedef __attribute__((ext_vector_type(8))) int i32x8;
typedef __attribute__((ext_vector_type(4))) int i32x4v;
// 1-D contiguous DMA: global f32[nelem] -> LDS byte offset (D# built per ISA 8.3/8.4)
__device__ __forceinline__ void tdm_load_1d(const float* gsrc, unsigned lds_byte_off, int nelem) {
  unsigned long ga = (unsigned long)(const void*)gsrc;
  u32x4 g0;
  g0[0] = 1u;                                             // count=1 valid descriptor
  g0[1] = lds_byte_off;                                   // lds_addr [63:32]
  g0[2] = (unsigned)(ga & 0xFFFFFFFFu);                   // global_addr[31:0]
  g0[3] = (unsigned)((ga >> 32) & 0x1FFFFFFu) | (2u << 30); // addr[56:32] | type=2
  i32x8 g1;
  g1[0] = 0x20000;                                        // data_size=4B (enc 2), mask=0
  g1[1] = (nelem & 0xFFFF) << 16;                         // tensor_dim0[15:0] @ bits 63:48
  g1[2] = ((nelem >> 16) & 0xFFFF) | (1 << 16);           // tensor_dim0 hi | tensor_dim1=1
  g1[3] = (nelem & 0xFFFF) << 16;                         // tile_dim0 @ bits 127:112
  g1[4] = 1;                                              // tile_dim1=1, tile_dim2=0
  g1[5] = nelem;                                          // tensor_dim0_stride[31:0]
  g1[6] = 0;
  g1[7] = 0;
  i32x4v z4 = {0, 0, 0, 0};
  i32x8  z8 = {0, 0, 0, 0, 0, 0, 0, 0};
  // 6-arg toolchain signature: (g0, g1, g2, g3, g4, cpol)
  __builtin_amdgcn_tensor_load_to_lds(g0, g1, z4, z4, z8, 0);
}
#else
#define USE_TDM 0
#endif

// ---------------- fast bf16 WMMA GEMM (M%128==0, N%64==0, K%32==0) ----------------
// 256 threads = 8 waves; block tile 128x64; each wave owns a 32x32 patch (4 wmma/K-step).
__global__ __launch_bounds__(256) void k_gemm_fast(const float* __restrict__ A, int lda,
    const float* __restrict__ Wt, int ldw, const float* __restrict__ bias,
    float* __restrict__ C, int ldc, int M, int N, int K, int flags)
{
  __shared__ __align__(16) unsigned short As[128][40];   // [m][k]
  __shared__ __align__(16) unsigned short Bs[64][40];    // [n][k] (K-transposed)
  const int tid  = threadIdx.x;
  const int wave = tid >> 5, lane = tid & 31;
  const int m0 = blockIdx.y * 128, n0 = blockIdx.x * 64;
  const int wr = (wave & 3) * 32;       // wave row base
  const int wc = (wave >> 2) * 32;      // wave col base
  v8f acc00 = {}, acc01 = {}, acc10 = {}, acc11 = {};
  // A stage: thread -> 16 contiguous elems of one row (128x32 tile)
  const int arow = tid >> 1, acol = (tid & 1) * 16;
  // B stage: thread -> 4 cols x 2 consecutive k rows (32x64 tile), packed pairwise
  const int bk2 = (tid >> 4) * 2, bnn = (tid & 15) * 4;
  const float* Ap = A  + (long)(m0 + arow) * lda + acol;
  const float* Wp = Wt + (long)bk2 * ldw + n0 + bnn;

  for (int k0 = 0; k0 < K; k0 += 32) {
    if (k0 + 32 < K) {                       // uniform-scalar branch; near-cache policy
      __builtin_prefetch(Ap + k0 + 32, 0, 3);
      __builtin_prefetch(Wp + (long)(k0 + 32) * ldw, 0, 3);
    }
    {
      const f32x4* p = reinterpret_cast<const f32x4*>(Ap + k0);
      f32x4 f0 = p[0], f1 = p[1], f2 = p[2], f3 = p[3];
      u32x4 o0, o1;
      o0[0]=pack2(f0[0],f0[1]); o0[1]=pack2(f0[2],f0[3]);
      o0[2]=pack2(f1[0],f1[1]); o0[3]=pack2(f1[2],f1[3]);
      o1[0]=pack2(f2[0],f2[1]); o1[1]=pack2(f2[2],f2[3]);
      o1[2]=pack2(f3[0],f3[1]); o1[3]=pack2(f3[2],f3[3]);
      *reinterpret_cast<u32x4*>(&As[arow][acol])     = o0;
      *reinterpret_cast<u32x4*>(&As[arow][acol + 8]) = o1;
    }
    {
      const f32x4 r0 = *reinterpret_cast<const f32x4*>(Wp + (long)k0 * ldw);
      const f32x4 r1 = *reinterpret_cast<const f32x4*>(Wp + (long)k0 * ldw + ldw);
      #pragma unroll
      for (int j = 0; j < 4; ++j)
        *reinterpret_cast<unsigned int*>(&Bs[bnn + j][bk2]) = pack2(r0[j], r1[j]);
    }
    __syncthreads();
    const int l15 = lane & 15, hh = lane >> 4;
    Frag af0, af1, bf0, bf1;
    // A 16-bit layout: elem i -> k = (lane>>4)*16 + i
    af0.q[0] = *reinterpret_cast<const u32x4*>(&As[wr      + l15][hh * 16]);
    af0.q[1] = *reinterpret_cast<const u32x4*>(&As[wr      + l15][hh * 16 + 8]);
    af1.q[0] = *reinterpret_cast<const u32x4*>(&As[wr + 16 + l15][hh * 16]);
    af1.q[1] = *reinterpret_cast<const u32x4*>(&As[wr + 16 + l15][hh * 16 + 8]);
    // B 16-bit layout: elem i -> k = (lane>>4)*16 + i
    bf0.q[0] = *reinterpret_cast<const u32x4*>(&Bs[wc      + l15][hh * 16]);
    bf0.q[1] = *reinterpret_cast<const u32x4*>(&Bs[wc      + l15][hh * 16 + 8]);
    bf1.q[0] = *reinterpret_cast<const u32x4*>(&Bs[wc + 16 + l15][hh * 16]);
    bf1.q[1] = *reinterpret_cast<const u32x4*>(&Bs[wc + 16 + l15][hh * 16 + 8]);
    acc00 = __builtin_amdgcn_wmma_f32_16x16x32_bf16(false, af0.v, false, bf0.v, (short)0, acc00, false, false);
    acc01 = __builtin_amdgcn_wmma_f32_16x16x32_bf16(false, af0.v, false, bf1.v, (short)0, acc01, false, false);
    acc10 = __builtin_amdgcn_wmma_f32_16x16x32_bf16(false, af1.v, false, bf0.v, (short)0, acc10, false, false);
    acc11 = __builtin_amdgcn_wmma_f32_16x16x32_bf16(false, af1.v, false, bf1.v, (short)0, acc11, false, false);
    __syncthreads();
  }
  const int l15 = lane & 15, hh = lane >> 4;
  #pragma unroll
  for (int i = 0; i < 2; ++i) {
    #pragma unroll
    for (int j = 0; j < 2; ++j) {
      v8f a = (i == 0) ? (j == 0 ? acc00 : acc01) : (j == 0 ? acc10 : acc11);
      int gn = n0 + wc + j * 16 + l15;
      float bsv = (flags & GF_BIAS) ? bias[gn] : 0.f;
      #pragma unroll
      for (int rr = 0; rr < 8; ++rr) {
        int gm = m0 + wr + i * 16 + rr + 8 * hh;   // C/D: m = vgpr + 8*(lane>>4)
        float v = a[rr] + bsv;
        if (flags & GF_ACCUM) v += C[(long)gm * ldc + gn];
        if (flags & GF_SILU)  v = siluf(v);
        if (flags & GF_SOFTPLUS) v = (v > 20.f) ? v : log1pf(__expf(v));
        C[(long)gm * ldc + gn] = v;
      }
    }
  }
}

// ---------------- safe GEMM (ragged shapes): branchless clamped loads ----------------
__global__ __launch_bounds__(256) void k_gemm_safe(const float* __restrict__ A, int lda,
    const float* __restrict__ Wt, int ldw, const float* __restrict__ bias,
    float* __restrict__ C, int ldc, int M, int N, int K, int flags)
{
  __shared__ __align__(16) unsigned short As[64][40];
  __shared__ __align__(16) unsigned short Bs[64][40];
  const int tid  = threadIdx.x;
  const int wave = tid >> 5, lane = tid & 31;
  const int m0 = blockIdx.y * 64, n0 = blockIdx.x * 64;
  const int rt = wave & 3, cp = wave >> 2;
  v8f acc0 = {}; v8f acc1 = {};
  const int arow = tid >> 2, acol = (tid & 3) * 8;
  const int bk   = tid >> 3, bn   = (tid & 7) * 8;
  const int gm   = m0 + arow;
  const int gmc  = (gm < M) ? gm : (M - 1);
  for (int k0 = 0; k0 < K; k0 += 32) {
    #pragma unroll
    for (int j = 0; j < 8; ++j) {           // clamped index + select-zero (no exec churn)
      int gk = k0 + acol + j;
      int gkc = (gk < K) ? gk : (K - 1);
      float f = A[(long)gmc * lda + gkc];
      if (gm >= M || gk >= K) f = 0.f;
      As[arow][acol + j] = (unsigned short)f2bf(f);
    }
    {
      int gk = k0 + bk;
      int gkc = (gk < K) ? gk : (K - 1);
      #pragma unroll
      for (int j = 0; j < 8; ++j) {
        int gn = n0 + bn + j;
        int gnc = (gn < N) ? gn : (N - 1);
        float f = Wt[(long)gkc * ldw + gnc];
        if (gk >= K || gn >= N) f = 0.f;
        Bs[bn + j][bk] = (unsigned short)f2bf(f);
      }
    }
    __syncthreads();
    const int l15 = lane & 15, hh = lane >> 4;
    Frag af;
    af.q[0] = *reinterpret_cast<const u32x4*>(&As[rt * 16 + l15][hh * 16]);
    af.q[1] = *reinterpret_cast<const u32x4*>(&As[rt * 16 + l15][hh * 16 + 8]);
    {
      Frag bf;
      int nc = (cp * 2 + 0) * 16 + l15;
      bf.q[0] = *reinterpret_cast<const u32x4*>(&Bs[nc][hh * 16]);
      bf.q[1] = *reinterpret_cast<const u32x4*>(&Bs[nc][hh * 16 + 8]);
      acc0 = __builtin_amdgcn_wmma_f32_16x16x32_bf16(false, af.v, false, bf.v, (short)0, acc0, false, false);
    }
    {
      Frag bf;
      int nc = (cp * 2 + 1) * 16 + l15;
      bf.q[0] = *reinterpret_cast<const u32x4*>(&Bs[nc][hh * 16]);
      bf.q[1] = *reinterpret_cast<const u32x4*>(&Bs[nc][hh * 16 + 8]);
      acc1 = __builtin_amdgcn_wmma_f32_16x16x32_bf16(false, af.v, false, bf.v, (short)0, acc1, false, false);
    }
    __syncthreads();
  }
  const int l15 = lane & 15, hh = lane >> 4;
  #pragma unroll
  for (int u = 0; u < 2; ++u) {
    int gn = n0 + (cp * 2 + u) * 16 + l15;
    #pragma unroll
    for (int rr = 0; rr < 8; ++rr) {
      int gmo = m0 + rt * 16 + rr + 8 * hh;
      if (gmo < M && gn < N) {
        float v = (u == 0) ? acc0[rr] : acc1[rr];
        if (flags & GF_BIAS)  v += bias[gn];
        if (flags & GF_ACCUM) v += C[(long)gmo * ldc + gn];
        if (flags & GF_SILU)  v = siluf(v);
        if (flags & GF_SOFTPLUS) v = (v > 20.f) ? v : log1pf(__expf(v));
        C[(long)gmo * ldc + gn] = v;
      }
    }
  }
}

// ---------------- elementwise / special kernels ----------------
__global__ void k_prep(const float* __restrict__ st, const float* __restrict__ dead,
                       float* __restrict__ ms, float* __restrict__ alive) {
  int i = blockIdx.x * 256 + threadIdx.x;
  if (i >= BT_TOK * 16) return;
  int tok = i >> 4, c = i & 15;
  float a = st[tok * 16] > 0.f ? 1.f : 0.f;
  ms[i] = (a > 0.f) ? st[i] : dead[c];
  if (c == 0) alive[tok] = a;
}

__global__ void k_addemb(float* __restrict__ se, const float* __restrict__ temb,
                         const float* __restrict__ shemb, const int* __restrict__ tid_) {
  long i = (long)blockIdx.x * 256 + threadIdx.x;
  if (i >= (long)BT_TOK * DM) return;
  int tok = (int)(i >> 9), c = (int)(i & 511);
  int n = tok & 7, b = tok >> 11;
  int team = tid_[b * NS + n];
  se[i] += temb[team * DM + c] + shemb[n * DM + c];
}

__global__ void k_relfeat(const float* __restrict__ pos, const float* __restrict__ vel,
                          float* __restrict__ feats) {
  int p = blockIdx.x * 256 + threadIdx.x;
  if (p >= BSNN) return;
  int m = p & 7, n = (p >> 3) & 7, bs = p >> 6;
  long pb = (long)bs * NS * 2;
  float dpx = (pos[pb + n*2]   - pos[pb + m*2])   * (1.f/1024.f);
  float dpy = (pos[pb + n*2+1] - pos[pb + m*2+1]) * (1.f/1024.f);
  float dvx =  vel[pb + n*2]   - vel[pb + m*2];
  float dvy =  vel[pb + n*2+1] - vel[pb + m*2+1];
  float dist = sqrtf(dpx*dpx + dpy*dpy);
  long o = (long)p * 6;
  feats[o]=dpx; feats[o+1]=dpy; feats[o+2]=dvx; feats[o+3]=dvy;
  feats[o+4]=dist; feats[o+5]=__expf(-dist);
}

__global__ __launch_bounds__(256) void k_rmsnorm(const float* __restrict__ in,
    const float* __restrict__ g, float* __restrict__ out, int dosilu) {
  __shared__ float red[256];
  long base = (long)blockIdx.x * DM;
  float ss = 0.f;
  for (int c = threadIdx.x; c < DM; c += 256) { float v = in[base + c]; ss += v * v; }
  red[threadIdx.x] = ss; __syncthreads();
  for (int st = 128; st > 0; st >>= 1) {
    if (threadIdx.x < st) red[threadIdx.x] += red[threadIdx.x + st];
    __syncthreads();
  }
  float inv = rsqrtf(red[0] * (1.f / DM) + 1e-6f);
  for (int c = threadIdx.x; c < DM; c += 256) {
    float v = in[base + c] * inv * g[c];
    if (dosilu) v = siluf(v);
    out[base + c] = v;
  }
}

__global__ void k_conv(const float* __restrict__ xz, const float* __restrict__ cw,
                       const float* __restrict__ cb, float* __restrict__ xc) {
  long i = (long)blockIdx.x * 256 + threadIdx.x;
  if (i >= (long)BT_TOK * DI) return;
  int tok = (int)(i >> 10), d = (int)(i & 1023);
  int n = tok & 7, s = (tok >> 3) & 255, b = tok >> 11;
  float acc = cb[d];
  #pragma unroll
  for (int j = 0; j < 4; ++j) {
    int sj = s - 3 + j;
    if (sj >= 0) {
      long t2 = (long)(b * SEQ + sj) * NS + n;
      acc += xz[t2 * 2048 + d] * cw[d * 4 + j];
    }
  }
  xc[i] = siluf(acc);
}

// selective scan: 64 (b,n) pairs x 4 channel-chunks of 256; h[16] in registers
__global__ __launch_bounds__(256) void k_scan(const float* __restrict__ xz,
    const float* __restrict__ xc, const float* __restrict__ xp,
    const float* __restrict__ dtb, const float* __restrict__ A_log,
    const float* __restrict__ Dp, float* __restrict__ y) {
  __shared__ float BC[32];
  int bn = blockIdx.x >> 2;
  int d  = ((blockIdx.x & 3) << 8) + threadIdx.x;
  int b = bn >> 3, n = bn & 7;
  float Av[DSTATE], h[DSTATE];
  #pragma unroll
  for (int k = 0; k < DSTATE; ++k) { Av[k] = -__expf(A_log[d * DSTATE + k]); h[k] = 0.f; }
  float Dd = Dp[d];
  for (int s = 0; s < SEQ; ++s) {
    long tok = (long)(b * SEQ + s) * NS + n;
    if (threadIdx.x < 32) BC[threadIdx.x] = xp[tok * 64 + 32 + threadIdx.x];
    __syncthreads();
    float dtv = dtb[tok * DI + d];
    float xv  = xc[tok * DI + d];
    float zv  = xz[tok * 2048 + DI + d];
    float acc = 0.f;
    #pragma unroll
    for (int k = 0; k < DSTATE; ++k) {
      h[k] = h[k] * __expf(dtv * Av[k]) + dtv * BC[k] * xv;
      acc += h[k] * BC[16 + k];
    }
    y[tok * DI + d] = (acc + xv * Dd) * siluf(zv);
    __syncthreads();
  }
}

// 8-entity multi-head attention, optional pair bias, key mask on !alive.
// Q/K/V slabs (16KB each, contiguous) staged into LDS via the Tensor Data Mover.
__global__ __launch_bounds__(256) void k_attn(const float* __restrict__ Q,
    const float* __restrict__ Kx, const float* __restrict__ Vx,
    const float* __restrict__ bias, const float* __restrict__ alive,
    float* __restrict__ O) {
  __shared__ __align__(16) float smem[3*NS*DM + 2*NH*NS*NS];
  float* qs = smem;
  float* ks = smem +     NS*DM;
  float* vs = smem + 2 * NS*DM;
  float* sc = smem + 3 * NS*DM;
  float* pr = sc + NH*NS*NS;
  int bs = blockIdx.x;
  long base = (long)bs * NS * DM;
#if USE_TDM
  if (threadIdx.x == 0) {                      // one wave issues 3 DMA descriptors
    tdm_load_1d(Q  + base, 0u,                    NS*DM);
    tdm_load_1d(Kx + base, (unsigned)(NS*DM*4),   NS*DM);
    tdm_load_1d(Vx + base, (unsigned)(2*NS*DM*4), NS*DM);
    __builtin_amdgcn_s_wait_tensorcnt(0);
  }
  __syncthreads();
#else
  for (int i = threadIdx.x; i < NS*DM; i += 256) {
    qs[i] = Q[base+i]; ks[i] = Kx[base+i]; vs[i] = Vx[base+i];
  }
  __syncthreads();
#endif
  for (int e = threadIdx.x; e < NH*NS*NS; e += 256) {
    int hh = e >> 6, n = (e >> 3) & 7, m = e & 7;
    const float* qp = &qs[n*DM + hh*DH];
    const float* kp = &ks[m*DM + hh*DH];
    float a = 0.f;
    #pragma unroll
    for (int dd = 0; dd < DH; ++dd) a += qp[dd]*kp[dd];
    a *= 0.125f;                                   // 1/sqrt(64)
    if (bias) a += bias[((long)bs*64 + n*8 + m)*8 + hh];
    if (alive[(long)bs*NS + m] < 0.5f) a = -1e9f;
    sc[e] = a;
  }
  __syncthreads();
  if (threadIdx.x < NH*NS) {
    int hh = threadIdx.x >> 3, n = threadIdx.x & 7;
    float* row  = &sc[hh*64 + n*8];
    float* prow = &pr[hh*64 + n*8];
    float mx = row[0];
    #pragma unroll
    for (int m = 1; m < 8; ++m) mx = fmaxf(mx, row[m]);
    float ssum = 0.f;
    #pragma unroll
    for (int m = 0; m < 8; ++m) { float e_ = __expf(row[m]-mx); prow[m]=e_; ssum += e_; }
    float inv = 1.f/ssum;
    #pragma unroll
    for (int m = 0; m < 8; ++m) prow[m] *= inv;
  }
  __syncthreads();
  for (int i = threadIdx.x; i < NS*DM; i += 256) {
    int n = i >> 9, c = i & 511, hh = c >> 6;
    float a = 0.f;
    #pragma unroll
    for (int m = 0; m < 8; ++m) a += pr[hh*64 + n*8 + m] * vs[m*DM + c];
    O[base + i] = a;
  }
}

__global__ void k_copy(const float* __restrict__ src, float* __restrict__ dst, long nElem) {
  long i = (long)blockIdx.x * 256 + threadIdx.x;
  if (i < nElem) dst[i] = src[i];
}

__global__ void k_shift(const float* __restrict__ xf, float* __restrict__ hist) {
  long i = (long)blockIdx.x * 256 + threadIdx.x;
  if (i >= (long)BT_TOK * DM) return;
  int tok = (int)(i >> 9);
  int s = (tok >> 3) & 255;
  hist[i] = (s > 0) ? xf[i - (long)NS * DM] : 0.f;
}

__global__ __launch_bounds__(256) void k_pool(const float* __restrict__ xe,
    const float* __restrict__ alive, float* __restrict__ pooled) {
  int bs = blockIdx.x;
  float asum = 0.f;
  #pragma unroll
  for (int n = 0; n < NS; ++n) asum += alive[bs*NS + n];
  float inv = 1.f / fmaxf(asum, 1.f);
  for (int c = threadIdx.x; c < DM; c += 256) {
    float acc = 0.f;
    #pragma unroll
    for (int n = 0; n < NS; ++n)
      acc += alive[bs*NS + n] * xe[((long)bs*NS + n)*DM + c];
    pooled[(long)bs*DM + c] = acc * inv;
  }
}

__global__ void k_rsum(const float* __restrict__ r4, float* __restrict__ out) {
  int i = blockIdx.x * 256 + threadIdx.x;
  if (i < BB*SEQ) out[i] = r4[i*4] + r4[i*4+1] + r4[i*4+2] + r4[i*4+3];
}

// ---------------- host side ----------------
struct MambaP { const float *A_log,*D,*conv_b,*conv_w,*dt_b,*dt_w,*in_b,*in_w,*out_b,*out_w,*xp_b,*xp_w; };
struct AttnP  { const float *kb,*kw,*ob,*ow,*qb,*qw,*vb,*vw; };
struct BlockP { AttnP at; MambaP mb; const float *n1,*n2; };

static inline void gemm(hipStream_t st, const float* A, int lda, const float* Wt, int ldw,
                        const float* bias, float* C, int ldc, int M, int N, int K, int flags) {
  if ((M % 128 == 0) && (N % 64 == 0) && (K % 32 == 0)) {
    dim3 g((unsigned)(N / 64), (unsigned)(M / 128));
    k_gemm_fast<<<g, 256, 0, st>>>(A, lda, Wt, ldw, bias, C, ldc, M, N, K, flags);
  } else {
    dim3 g((unsigned)CDIV(N, 64), (unsigned)CDIV(M, 64));
    k_gemm_safe<<<g, 256, 0, st>>>(A, lda, Wt, ldw, bias, C, ldc, M, N, K, flags);
  }
}

extern "C" void kernel_launch(void* const* d_in, const int* in_sizes, int n_in,
                              void* d_out, int out_size, void* d_ws, size_t ws_size,
                              hipStream_t stream) {
  (void)in_sizes; (void)n_in; (void)out_size; (void)ws_size;
  // ---- raw inputs ----
  const float* state       = (const float*)d_in[0];
  const float* prev_action = (const float*)d_in[1];
  const float* pos         = (const float*)d_in[2];
  const float* vel         = (const float*)d_in[3];
  const int*   team_ids    = (const int*)d_in[4];
  // ---- params: jax pytree flatten order (dict keys sorted) ----
  int pi = 5;
  auto nxt = [&]() { return (const float*)d_in[pi++]; };
  const float* act_enc_b = nxt(); const float* act_enc_w = nxt();
  AttnP aat; aat.kb=nxt(); aat.kw=nxt(); aat.ob=nxt(); aat.ow=nxt();
             aat.qb=nxt(); aat.qw=nxt(); aat.vb=nxt(); aat.vw=nxt();
  const float *ah0b=nxt(), *ah0w=nxt(), *ah1b=nxt(), *ah1w=nxt();
  const float *sp_norm=nxt(), *tmp_norm=nxt();
  const float *adb[5], *adw[5];
  for (int i = 0; i < 5; ++i) { adb[i]=nxt(); adw[i]=nxt(); }
  BlockP blk[4];
  for (int i = 0; i < 4; ++i) {
    blk[i].at.kb=nxt(); blk[i].at.kw=nxt(); blk[i].at.ob=nxt(); blk[i].at.ow=nxt();
    blk[i].at.qb=nxt(); blk[i].at.qw=nxt(); blk[i].at.vb=nxt(); blk[i].at.vw=nxt();
    blk[i].mb.A_log=nxt(); blk[i].mb.D=nxt(); blk[i].mb.conv_b=nxt(); blk[i].mb.conv_w=nxt();
    blk[i].mb.dt_b=nxt();  blk[i].mb.dt_w=nxt(); blk[i].mb.in_b=nxt(); blk[i].mb.in_w=nxt();
    blk[i].mb.out_b=nxt(); blk[i].mb.out_w=nxt(); blk[i].mb.xp_b=nxt(); blk[i].mb.xp_w=nxt();
    blk[i].n1=nxt(); blk[i].n2=nxt();
  }
  const float* dead = nxt();
  const float *fus_g=nxt(), *fus_b=nxt(), *fus_w=nxt();
  const float *rt0b=nxt(), *rt0w=nxt(), *rt1b=nxt(), *rt1w=nxt();
  const float* ship_emb = nxt();
  const float *se0b=nxt(), *se0w=nxt(), *se1b=nxt(), *se1w=nxt();
  const float* team_emb = nxt();
  const float *te_pre_b=nxt(), *te_pre_w=nxt(), *te_rw_b=nxt(), *te_rw_w=nxt(),
              *te_v_b=nxt(),  *te_v_w=nxt();
  AttnP tat; tat.kb=nxt(); tat.kw=nxt(); tat.ob=nxt(); tat.ow=nxt();
             tat.qb=nxt(); tat.qw=nxt(); tat.vb=nxt(); tat.vw=nxt();
  const float *wh0b=nxt(), *wh0w=nxt(), *wh1b=nxt(), *wh1w=nxt();

  // ---- outputs (concatenated flat, return order) ----
  float* out_state  = (float*)d_out;                          // [B,S,N,16]
  float* out_logit  = out_state  + (long)BT_TOK*16;           // [B,S,N,12]
  float* out_value  = out_logit  + (long)BT_TOK*12;           // [B,S,1]
  float* out_reward = out_value  + BB*SEQ;                    // [B,S,1]
  float* out_xf     = out_reward + BB*SEQ;                    // [B,S,N,512] == x

  // ---- workspace layout (floats); aliases reuse dead buffers ----
  float* Wf = (float*)d_ws;
  size_t off = 0;
  auto alloc = [&](size_t n) { float* r = Wf + off; off += n; return r; };
  float* ms    = alloc((size_t)BT_TOK*16);
  float* alive = alloc((size_t)BT_TOK);
  float* semb  = alloc((size_t)BT_TOK*DM);
  float* t0    = alloc((size_t)BT_TOK*DM);
  float* tn    = alloc((size_t)BT_TOK*DM);
  float* xzb   = alloc((size_t)BT_TOK*2048);
  float* xcb   = alloc((size_t)BT_TOK*DI);
  float* xpb   = alloc((size_t)BT_TOK*64);
  float* dtbuf = alloc((size_t)BT_TOK*DI);
  float* ybuf  = alloc((size_t)BT_TOK*DI);
  float* trunk = alloc((size_t)BSNN*64);
  float* biasb = alloc((size_t)BSNN*8);
  float* bq    = alloc((size_t)BT_TOK*DM);
  float* bk    = alloc((size_t)BT_TOK*DM);
  float* bv    = alloc((size_t)BT_TOK*DM);
  float* bao   = alloc((size_t)BT_TOK*DM);
  float* poolb = alloc((size_t)BB*SEQ*DM);
  float* r4    = alloc((size_t)BB*SEQ*4);
  float* aemb  = xcb;    // free until first conv
  float* feats = xzb;    // free until first in_proj
  float* rtb   = t0;     // trunk temp, t0 reused for heads later
  float* hist  = dtbuf;  // free after layers
  float* xas   = ybuf;   // free after layers
  float* xe    = bq;     // free after temporal attention
  float* x     = out_xf; // residual stream lives in the output segment

  const long TD = (long)BT_TOK * DM;

  // ---- embed ----
  k_prep<<<CDIV(BT_TOK*16,256),256,0,stream>>>(state, dead, ms, alive);
  gemm(stream, ms,16,  se0w,DM, se0b, t0,DM,   BT_TOK,DM,16, GF_BIAS|GF_SILU);
  gemm(stream, t0,DM,  se1w,DM, se1b, semb,DM, BT_TOK,DM,DM, GF_BIAS);
  k_addemb<<<CDIV(TD,256),256,0,stream>>>(semb, team_emb, ship_emb, team_ids);
  gemm(stream, prev_action,12, act_enc_w,128, act_enc_b, aemb,128, BT_TOK,128,12, GF_BIAS|GF_SILU);
  gemm(stream, semb,DM,  fus_w,DM,                 fus_b,   x,DM, BT_TOK,DM,DM,  GF_BIAS);
  gemm(stream, aemb,128, fus_w + (size_t)DM*DM,DM, nullptr, x,DM, BT_TOK,DM,128, GF_ACCUM);
  k_rmsnorm<<<BT_TOK,256,0,stream>>>(x, fus_g, x, 1);

  // ---- relational trunk ----
  k_relfeat<<<CDIV(BSNN,256),256,0,stream>>>(pos, vel, feats);
  gemm(stream, feats,6, rt0w,64, rt0b, rtb,64,   BSNN,64,6,  GF_BIAS|GF_SILU);
  gemm(stream, rtb,64,  rt1w,64, rt1b, trunk,64, BSNN,64,64, GF_BIAS|GF_SILU);

  // ---- mamba + relational attention layers ----
  for (int i = 0; i < 4; ++i) {
    const BlockP& L = blk[i];
    k_rmsnorm<<<BT_TOK,256,0,stream>>>(x, L.n1, tn, 0);
    gemm(stream, tn,DM,  L.mb.in_w,2048, L.mb.in_b, xzb,2048, BT_TOK,2048,DM, GF_BIAS);
    k_conv<<<CDIV((long)BT_TOK*DI,256),256,0,stream>>>(xzb, L.mb.conv_w, L.mb.conv_b, xcb);
    gemm(stream, xcb,DI, L.mb.xp_w,64,  L.mb.xp_b, xpb,64,   BT_TOK,64,DI,     GF_BIAS);
    gemm(stream, xpb,64, L.mb.dt_w,DI,  L.mb.dt_b, dtbuf,DI, BT_TOK,DI,DTRANK, GF_BIAS|GF_SOFTPLUS);
    k_scan<<<256,256,0,stream>>>(xzb, xcb, xpb, dtbuf, L.mb.A_log, L.mb.D, ybuf);
    gemm(stream, ybuf,DI, L.mb.out_w,DM, L.mb.out_b, x,DM, BT_TOK,DM,DI, GF_BIAS|GF_ACCUM);
    k_rmsnorm<<<BT_TOK,256,0,stream>>>(x, L.n2, tn, 0);
    gemm(stream, trunk,64, adw[i+1],8, adb[i+1], biasb,8, BSNN,8,64, GF_BIAS);
    gemm(stream, tn,DM, L.at.qw,DM, L.at.qb, bq,DM, BT_TOK,DM,DM, GF_BIAS);
    gemm(stream, tn,DM, L.at.kw,DM, L.at.kb, bk,DM, BT_TOK,DM,DM, GF_BIAS);
    gemm(stream, tn,DM, L.at.vw,DM, L.at.vb, bv,DM, BT_TOK,DM,DM, GF_BIAS);
    k_attn<<<BB*SEQ,256,0,stream>>>(bq, bk, bv, biasb, alive, bao);
    gemm(stream, bao,DM, L.at.ow,DM, L.at.ob, x,DM, BT_TOK,DM,DM, GF_BIAS|GF_ACCUM);
  }

  // ---- world head (x == x_final, already in d_out) ----
  gemm(stream, x,DM,  wh0w,DM, wh0b, t0,DM,        BT_TOK,DM,DM, GF_BIAS|GF_SILU);
  gemm(stream, t0,DM, wh1w,16, wh1b, out_state,16, BT_TOK,16,DM, GF_BIAS);

  // ---- actor spatial attention ----
  gemm(stream, trunk,64, adw[0],8, adb[0], biasb,8, BSNN,8,64, GF_BIAS);
  k_rmsnorm<<<BT_TOK,256,0,stream>>>(semb, sp_norm, tn, 0);
  gemm(stream, tn,DM, aat.qw,DM, aat.qb, bq,DM, BT_TOK,DM,DM, GF_BIAS);
  gemm(stream, tn,DM, aat.kw,DM, aat.kb, bk,DM, BT_TOK,DM,DM, GF_BIAS);
  gemm(stream, tn,DM, aat.vw,DM, aat.vb, bv,DM, BT_TOK,DM,DM, GF_BIAS);
  k_attn<<<BB*SEQ,256,0,stream>>>(bq, bk, bv, biasb, alive, bao);
  k_copy<<<CDIV(TD,256),256,0,stream>>>(semb, xas, TD);
  gemm(stream, bao,DM, aat.ow,DM, aat.ob, xas,DM, BT_TOK,DM,DM, GF_BIAS|GF_ACCUM);

  // ---- actor temporal attention (kv = shifted history of x_final) ----
  k_shift<<<CDIV(TD,256),256,0,stream>>>(x, hist);
  k_rmsnorm<<<BT_TOK,256,0,stream>>>(xas, tmp_norm, tn, 0);
  gemm(stream, tn,DM,   tat.qw,DM, tat.qb, bq,DM, BT_TOK,DM,DM, GF_BIAS);
  gemm(stream, hist,DM, tat.kw,DM, tat.kb, bk,DM, BT_TOK,DM,DM, GF_BIAS);
  gemm(stream, hist,DM, tat.vw,DM, tat.vb, bv,DM, BT_TOK,DM,DM, GF_BIAS);
  k_attn<<<BB*SEQ,256,0,stream>>>(bq, bk, bv, nullptr, alive, bao);
  gemm(stream, bao,DM, tat.ow,DM, tat.ob, xas,DM, BT_TOK,DM,DM, GF_BIAS|GF_ACCUM);

  // ---- actor head ----
  gemm(stream, xas,DM, ah0w,DM, ah0b, t0,DM,        BT_TOK,DM,DM, GF_BIAS|GF_SILU);
  gemm(stream, t0,DM,  ah1w,12, ah1b, out_logit,12, BT_TOK,12,DM, GF_BIAS);

  // ---- team eval ----
  gemm(stream, xas,DM, te_pre_w,DM, te_pre_b, xe,DM, BT_TOK,DM,DM, GF_BIAS|GF_SILU);
  k_pool<<<BB*SEQ,256,0,stream>>>(xe, alive, poolb);
  gemm(stream, poolb,DM, te_v_w,1,  te_v_b,  out_value,1, BB*SEQ,1,DM, GF_BIAS);
  gemm(stream, poolb,DM, te_rw_w,4, te_rw_b, r4,4,        BB*SEQ,4,DM, GF_BIAS);
  k_rsum<<<CDIV(BB*SEQ,256),256,0,stream>>>(r4, out_reward);
}
